// RelPartialLearnableMultiHeadAttn_41901700940373
// MI455X (gfx1250) — matrix operand order repaired and checked by reading, not compiled, not run on hardware
//
#include <hip/hip_runtime.h>
#include <hip/hip_bf16.h>

// ---------------------------------------------------------------------------
// Transformer-XL relative multi-head attention for gfx1250 (MI455X).
// N_HEAD=16, D_HEAD=64, D_MODEL=1024, QLEN=1024, MLEN=1024, KLEN=2048, BSZ=4
// bf16 WMMA everywhere; attention tiled per (b, head, 16 q-rows) with full
// 2048-wide score rows resident in LDS (CDNA5 WGP has 320KB). rel_shift fused
// via diagonal read of raw-BD LDS buffer. GEMM k-loops software-pipelined
// (register double-buffering, no-unroll so the 1-iteration slack survives).
// ---------------------------------------------------------------------------

typedef __bf16 bf16_t;
typedef __attribute__((ext_vector_type(16))) __bf16 v16bf;
typedef __attribute__((ext_vector_type(8)))  __bf16 v8bf;   // 16 bytes
typedef __attribute__((ext_vector_type(8)))  float  v8f;

#define N_HEAD  16
#define D_HEAD  64
#define D_MODEL 1024
#define QLEN    1024
#define MLEN    1024
#define KLEN    2048
#define BSZ     4
#define SCALE_F 0.125f   // 1/sqrt(64)

union Frag16 { v16bf v; v8bf h[2]; };

// A-operand fragment (16x32 bf16, MxK), A row-major [*, lda], K contiguous.
// lanes 0-15: row=lane,  K = k0+{0..7}, k0+{16..23}
// lanes16-31: row=lane-16,K = k0+{8..15}, k0+{24..31}
__device__ __forceinline__ v16bf load_a_frag(const bf16_t* A, int lda,
                                             int row_base, int k0, int lane) {
    int half = lane >> 4;
    int row  = row_base + (lane & 15);
    const bf16_t* p = A + (size_t)row * lda + k0 + half * 8;
    Frag16 f;
    f.h[0] = *(const v8bf*)(p);
    f.h[1] = *(const v8bf*)(p + 16);
    return f.v;
}

// B-operand fragment (32x16 bf16, KxN) sourced from row-major [N, ldb] (i.e. Bt),
// K contiguous within a row.
// lanes 0-15: col=lane,   K = k0+{0..7}, k0+{16..23}
// lanes16-31: col=lane-16,K = k0+{8..15}, k0+{24..31}
__device__ __forceinline__ v16bf load_b_frag(const bf16_t* Bt, int ldb,
                                             int col_base, int k0, int lane) {
    int row = col_base + (lane & 15);
    const bf16_t* p = Bt + (size_t)row * ldb + k0 + ((lane >> 4) << 3);
    Frag16 f;
    f.h[0] = *(const v8bf*)(p);
    f.h[1] = *(const v8bf*)(p + 16);
    return f.v;
}

__device__ __forceinline__ v8f wmma_bf16(v16bf a, v16bf b, v8f c) {
    return __builtin_amdgcn_wmma_f32_16x16x32_bf16(false, a, false, b,
                                                   (short)0, c, false, false);
}

// ---------------------------------------------------------------------------
// K0: f32 -> bf16 convert
// ---------------------------------------------------------------------------
__global__ void cvt_f32_bf16(const float* __restrict__ in,
                             bf16_t* __restrict__ out, int n) {
    int i = blockIdx.x * blockDim.x + threadIdx.x;
    if (i < n) out[i] = (bf16_t)in[i];
}

// ---------------------------------------------------------------------------
// K1/K2/K4: generic WMMA GEMM  C[M,N] = A[M,K] @ B[N,K]^T  with epilogues.
// Block: 256 thr = 8 waves as 4(M) x 2(N); block tile 64(M) x 128(N);
// wave tile 16(M) x 64(N) = 4 accumulators. k-loop software pipelined.
// ---------------------------------------------------------------------------
#define EP_QKV 0
#define EP_RK  1
#define EP_F32 2

__global__ void wmma_gemm_kernel(const bf16_t* __restrict__ A,
                                 const bf16_t* __restrict__ B,
                                 int M, int N, int K, int mode,
                                 bf16_t* __restrict__ qw, bf16_t* __restrict__ qr,
                                 bf16_t* __restrict__ kk, bf16_t* __restrict__ vt,
                                 const float* __restrict__ rwb,
                                 const float* __restrict__ rrb,
                                 bf16_t* __restrict__ rk,
                                 float* __restrict__ outp) {
    int tid  = threadIdx.x;
    int lane = tid & 31;
    int wave = tid >> 5;
    int wm   = wave & 3;       // 0..3
    int wn   = wave >> 2;      // 0..1
    int m0 = blockIdx.y * 64 + wm * 16;
    int n0 = blockIdx.x * 128 + wn * 64;

    v8f acc[4];
    #pragma unroll
    for (int t = 0; t < 4; ++t) acc[t] = {};

    // ---- software-pipelined k-loop: fragments for step s+1 load while step s
    // ---- feeds the matrix pipes. no-unroll keeps the one-iteration slack so
    // ---- the s_wait_loadcnt before each WMMA covers only last iteration's loads.
    v16bf a_cur  = load_a_frag(A, K, m0, 0, lane);
    v16bf b_cur0 = load_b_frag(B, K, n0 +  0, 0, lane);
    v16bf b_cur1 = load_b_frag(B, K, n0 + 16, 0, lane);
    v16bf b_cur2 = load_b_frag(B, K, n0 + 32, 0, lane);
    v16bf b_cur3 = load_b_frag(B, K, n0 + 48, 0, lane);

    const int ksteps = K >> 5;
    #pragma unroll 1
    for (int s = 0; s < ksteps - 1; ++s) {
        int kn = (s + 1) << 5;
        // prefetch A one step further ahead into L2/WGP$ (global_prefetch_b8)
        __builtin_prefetch(A + (size_t)(m0 + (lane & 15)) * K + kn + 32, 0, 1);
        v16bf a_nxt  = load_a_frag(A, K, m0, kn, lane);
        v16bf b_nxt0 = load_b_frag(B, K, n0 +  0, kn, lane);
        v16bf b_nxt1 = load_b_frag(B, K, n0 + 16, kn, lane);
        v16bf b_nxt2 = load_b_frag(B, K, n0 + 32, kn, lane);
        v16bf b_nxt3 = load_b_frag(B, K, n0 + 48, kn, lane);
        acc[0] = wmma_bf16(a_cur, b_cur0, acc[0]);
        acc[1] = wmma_bf16(a_cur, b_cur1, acc[1]);
        acc[2] = wmma_bf16(a_cur, b_cur2, acc[2]);
        acc[3] = wmma_bf16(a_cur, b_cur3, acc[3]);
        a_cur = a_nxt;
        b_cur0 = b_nxt0; b_cur1 = b_nxt1; b_cur2 = b_nxt2; b_cur3 = b_nxt3;
    }
    acc[0] = wmma_bf16(a_cur, b_cur0, acc[0]);
    acc[1] = wmma_bf16(a_cur, b_cur1, acc[1]);
    acc[2] = wmma_bf16(a_cur, b_cur2, acc[2]);
    acc[3] = wmma_bf16(a_cur, b_cur3, acc[3]);

    // Epilogue. C layout: lane<16 -> rows m0+r (r=0..7), col n0+16t+lane;
    //                     lane>=16 -> rows m0+8+r, col n0+16t+(lane-16)
    int rofs = (lane >= 16) ? 8 : 0;
    #pragma unroll
    for (int t = 0; t < 4; ++t) {
        int col = n0 + 16 * t + (lane & 15);
        #pragma unroll
        for (int r = 0; r < 8; ++r) {
            int row = m0 + r + rofs;
            float v = acc[t][r];
            if (mode == EP_QKV) {
                // A rows are cat tokens: row = tkn*BSZ + b, cols: [q|k|v] x (h,d)
                int tkn = row >> 2, b = row & 3;
                int sec = col >> 10;          // 0=q,1=k,2=v
                int hc  = col & 1023;
                int h = hc >> 6, d = hc & 63;
                if (sec == 0) {
                    if (tkn >= MLEN) {
                        int i = tkn - MLEN;
                        size_t o = ((size_t)(b * N_HEAD + h) * QLEN + i) * D_HEAD + d;
                        qw[o] = (bf16_t)(v + rwb[hc]);
                        qr[o] = (bf16_t)(v + rrb[hc]);
                    }
                } else if (sec == 1) {
                    kk[((size_t)(b * N_HEAD + h) * KLEN + tkn) * D_HEAD + d] = (bf16_t)v;
                } else {
                    vt[((size_t)(b * N_HEAD + h) * D_HEAD + d) * KLEN + tkn] = (bf16_t)v;
                }
            } else if (mode == EP_RK) {
                // Rk[h][c][d], rows of A are c (0..KLEN-1), col = h*64+d
                int h = col >> 6, d = col & 63;
                rk[((size_t)h * KLEN + row) * D_HEAD + d] = (bf16_t)v;
            } else {
                outp[(size_t)row * N + col] = v;
            }
        }
    }
}

// ---------------------------------------------------------------------------
// K3: attention per (qtile, head, b). 128 threads = 4 waves; each wave owns
// 512 score columns. LDS: S f32[16][2048] (128KB) + Braw bf16[16][2048]
// (64KB, reused as P) + rowinv f32[16].
// ---------------------------------------------------------------------------
__global__ void attn_kernel(const bf16_t* __restrict__ Qw,
                            const bf16_t* __restrict__ Qr,
                            const bf16_t* __restrict__ Kb,
                            const bf16_t* __restrict__ Vt,
                            const bf16_t* __restrict__ Rk,
                            bf16_t* __restrict__ attnbf) {
    extern __shared__ float smem[];
    float*  S  = smem;                                 // [16][2048] f32
    bf16_t* Bb = (bf16_t*)(smem + 16 * KLEN);          // [16][2048] bf16
    float*  rowinv = (float*)(Bb + 16 * KLEN);         // [16]

    int tid  = threadIdx.x;
    int lane = tid & 31;
    int w    = tid >> 5;       // wave 0..3
    int i0   = blockIdx.x * 16;          // q tile base
    int n    = blockIdx.y;               // head
    int b    = blockIdx.z;               // batch

    const bf16_t* Qwb = Qw + ((size_t)(b * N_HEAD + n) * QLEN + i0) * D_HEAD;
    const bf16_t* Qrb = Qr + ((size_t)(b * N_HEAD + n) * QLEN + i0) * D_HEAD;
    const bf16_t* Kbb = Kb + (size_t)(b * N_HEAD + n) * KLEN * D_HEAD;
    const bf16_t* Vtb = Vt + (size_t)(b * N_HEAD + n) * D_HEAD * KLEN;
    const bf16_t* Rkb = Rk + (size_t)n * KLEN * D_HEAD;

    // Q fragments for d=0..31 and d=32..63
    v16bf qwf0 = load_a_frag(Qwb, D_HEAD, 0, 0,  lane);
    v16bf qwf1 = load_a_frag(Qwb, D_HEAD, 0, 32, lane);
    v16bf qrf0 = load_a_frag(Qrb, D_HEAD, 0, 0,  lane);
    v16bf qrf1 = load_a_frag(Qrb, D_HEAD, 0, 32, lane);

    int rofs = (lane >= 16) ? 8 : 0;
    int jbase = w * 512;

    // Phase 1: AC and raw BD tiles -> LDS (wave w covers cols [jbase, jbase+512)).
    // Software-pipelined; last tile peeled so in-loop loads are unconditional.
    v16bf bk0 = load_b_frag(Kbb, D_HEAD, jbase, 0,  lane);
    v16bf bk1 = load_b_frag(Kbb, D_HEAD, jbase, 32, lane);
    v16bf br0 = load_b_frag(Rkb, D_HEAD, jbase, 0,  lane);
    v16bf br1 = load_b_frag(Rkb, D_HEAD, jbase, 32, lane);
    #pragma unroll 1
    for (int jt = 0; jt < 31; ++jt) {
        int jn = jbase + (jt + 1) * 16;
        v16bf nk0 = load_b_frag(Kbb, D_HEAD, jn, 0,  lane);
        v16bf nk1 = load_b_frag(Kbb, D_HEAD, jn, 32, lane);
        v16bf nr0 = load_b_frag(Rkb, D_HEAD, jn, 0,  lane);
        v16bf nr1 = load_b_frag(Rkb, D_HEAD, jn, 32, lane);
        v8f ac = {};
        v8f bd = {};
        ac = wmma_bf16(qwf0, bk0, ac);
        ac = wmma_bf16(qwf1, bk1, ac);
        bd = wmma_bf16(qrf0, br0, bd);
        bd = wmma_bf16(qrf1, br1, bd);
        int col = jbase + jt * 16 + (lane & 15);
        #pragma unroll
        for (int r = 0; r < 8; ++r) {
            int row = r + rofs;
            S [row * KLEN + col] = ac[r];
            Bb[row * KLEN + col] = (bf16_t)bd[r];
        }
        bk0 = nk0; bk1 = nk1; br0 = nr0; br1 = nr1;
    }
    {   // jt = 31 (peeled)
        v8f ac = {};
        v8f bd = {};
        ac = wmma_bf16(qwf0, bk0, ac);
        ac = wmma_bf16(qwf1, bk1, ac);
        bd = wmma_bf16(qrf0, br0, bd);
        bd = wmma_bf16(qrf1, br1, bd);
        int col = jbase + 31 * 16 + (lane & 15);
        #pragma unroll
        for (int r = 0; r < 8; ++r) {
            int row = r + rofs;
            S [row * KLEN + col] = ac[r];
            Bb[row * KLEN + col] = (bf16_t)bd[r];
        }
    }
    __syncthreads();

    // Phase 2: fold rel-shifted BD, scale, causal mask.
    // shifted BD[i][j] = raw[i][j + QLEN-1 - qi], OOB -> 0
    for (int idx = tid; idx < 16 * KLEN; idx += 128) {
        int i = idx >> 11, j = idx & (KLEN - 1);
        int qi = i0 + i;
        int c = j + (QLEN - 1) - qi;
        float bdv = (c < KLEN) ? (float)Bb[i * KLEN + c] : 0.0f;
        float v = (S[idx] + bdv) * SCALE_F;
        if (j > qi + MLEN) v = -1e30f;
        S[idx] = v;
    }
    __syncthreads();

    // Phase 3: row softmax (wave w handles rows w*4..w*4+3); write P bf16 over Bb.
    for (int rr = 0; rr < 4; ++rr) {
        int i = w * 4 + rr;
        float m = -3.0e38f;
        for (int j = lane; j < KLEN; j += 32) m = fmaxf(m, S[i * KLEN + j]);
        #pragma unroll
        for (int off = 16; off > 0; off >>= 1) m = fmaxf(m, __shfl_xor(m, off, 32));
        float s = 0.0f;
        for (int j = lane; j < KLEN; j += 32) {
            float p = __expf(S[i * KLEN + j] - m);
            s += p;
            Bb[i * KLEN + j] = (bf16_t)p;
        }
        #pragma unroll
        for (int off = 16; off > 0; off >>= 1) s += __shfl_xor(s, off, 32);
        if (lane == 0) rowinv[i] = 1.0f / s;
    }
    __syncthreads();

    // Phase 4: attn_vec[16, 64] = P[16,2048] @ V[2048,64]; wave w owns d-cols
    // w*16..+15. Pipeline the global Vt fragments; P comes from LDS.
    int d0 = w * 16;
    v8f av = {};
    int half = lane >> 4;
    int prow = lane & 15;
    v16bf bv = load_b_frag(Vtb, KLEN, d0, 0, lane);
    #pragma unroll 1
    for (int k0 = 0; k0 < KLEN - 32; k0 += 32) {
        v16bf bvn = load_b_frag(Vtb, KLEN, d0, k0 + 32, lane);
        Frag16 fa;
        const bf16_t* pa = Bb + (size_t)prow * KLEN + k0 + half * 8;
        fa.h[0] = *(const v8bf*)(pa);
        fa.h[1] = *(const v8bf*)(pa + 16);
        av = wmma_bf16(fa.v, bv, av);
        bv = bvn;
    }
    {   // last k-step peeled
        Frag16 fa;
        const bf16_t* pa = Bb + (size_t)prow * KLEN + (KLEN - 32) + half * 8;
        fa.h[0] = *(const v8bf*)(pa);
        fa.h[1] = *(const v8bf*)(pa + 16);
        av = wmma_bf16(fa.v, bv, av);
    }
    int col = d0 + (lane & 15);
    #pragma unroll
    for (int r = 0; r < 8; ++r) {
        int i = r + rofs;
        float v = av[r] * rowinv[i];
        attnbf[((size_t)(i0 + i) * BSZ + b) * (N_HEAD * D_HEAD) + n * D_HEAD + col] =
            (bf16_t)v;
    }
}

// ---------------------------------------------------------------------------
// Host-side launcher
// ---------------------------------------------------------------------------
extern "C" void kernel_launch(void* const* d_in, const int* in_sizes, int n_in,
                              void* d_out, int out_size, void* d_ws, size_t ws_size,
                              hipStream_t stream) {
    const float* w     = (const float*)d_in[0];   // [1024,4,1024]
    const float* r     = (const float*)d_in[1];   // [2048,1024]
    const float* mems  = (const float*)d_in[2];   // [1024,4,1024]
    const float* rwb   = (const float*)d_in[3];   // [16,64]
    const float* rrb   = (const float*)d_in[4];   // [16,64]
    const float* qkvw  = (const float*)d_in[5];   // [3072,1024]
    const float* rnetw = (const float*)d_in[6];   // [1024,1024]
    const float* ow    = (const float*)d_in[7];   // [1024,1024]
    float* out = (float*)d_out;                   // [1024,4,1024]

    char* ws = (char*)d_ws;
    size_t off = 0;
    auto alloc = [&](size_t bytes) { char* p = ws + off; off += (bytes + 255) & ~(size_t)255; return p; };

    bf16_t* catbf  = (bf16_t*)alloc((size_t)KLEN * BSZ * D_MODEL * 2);      // [klen*b, 1024]
    bf16_t* rbf    = (bf16_t*)alloc((size_t)KLEN * D_MODEL * 2);
    bf16_t* qkvwbf = (bf16_t*)alloc((size_t)3 * D_MODEL * D_MODEL * 2);
    bf16_t* rnetbf = (bf16_t*)alloc((size_t)D_MODEL * D_MODEL * 2);
    bf16_t* owbf   = (bf16_t*)alloc((size_t)D_MODEL * D_MODEL * 2);
    bf16_t* Qw     = (bf16_t*)alloc((size_t)BSZ * N_HEAD * QLEN * D_HEAD * 2);
    bf16_t* Qr     = (bf16_t*)alloc((size_t)BSZ * N_HEAD * QLEN * D_HEAD * 2);
    bf16_t* Kb     = (bf16_t*)alloc((size_t)BSZ * N_HEAD * KLEN * D_HEAD * 2);
    bf16_t* Vt     = (bf16_t*)alloc((size_t)BSZ * N_HEAD * D_HEAD * KLEN * 2);
    bf16_t* Rk     = (bf16_t*)alloc((size_t)N_HEAD * KLEN * D_HEAD * 2);
    bf16_t* attnbf = (bf16_t*)alloc((size_t)QLEN * BSZ * N_HEAD * D_HEAD * 2);

    // K0: conversions (cat = [mems; w] along time axis -> contiguous concat)
    const int memsN = MLEN * BSZ * D_MODEL;
    const int wN    = QLEN * BSZ * D_MODEL;
    cvt_f32_bf16<<<(memsN + 255) / 256, 256, 0, stream>>>(mems, catbf, memsN);
    cvt_f32_bf16<<<(wN + 255) / 256, 256, 0, stream>>>(w, catbf + memsN, wN);
    cvt_f32_bf16<<<(KLEN * D_MODEL + 255) / 256, 256, 0, stream>>>(r, rbf, KLEN * D_MODEL);
    cvt_f32_bf16<<<(3 * D_MODEL * D_MODEL + 255) / 256, 256, 0, stream>>>(qkvw, qkvwbf, 3 * D_MODEL * D_MODEL);
    cvt_f32_bf16<<<(D_MODEL * D_MODEL + 255) / 256, 256, 0, stream>>>(rnetw, rnetbf, D_MODEL * D_MODEL);
    cvt_f32_bf16<<<(D_MODEL * D_MODEL + 255) / 256, 256, 0, stream>>>(ow, owbf, D_MODEL * D_MODEL);

    // K1: qkv projection  [8192,1024] @ [3072,1024]^T with head scatter epilogue
    {
        dim3 grid((3 * D_MODEL) / 128, (KLEN * BSZ) / 64);
        wmma_gemm_kernel<<<grid, 256, 0, stream>>>(catbf, qkvwbf,
            KLEN * BSZ, 3 * D_MODEL, D_MODEL, EP_QKV,
            Qw, Qr, Kb, Vt, rwb, rrb, nullptr, nullptr);
    }
    // K2: r projection  [2048,1024] @ [1024,1024]^T -> Rk[h][c][d]
    {
        dim3 grid(D_MODEL / 128, KLEN / 64);
        wmma_gemm_kernel<<<grid, 256, 0, stream>>>(rbf, rnetbf,
            KLEN, D_MODEL, D_MODEL, EP_RK,
            nullptr, nullptr, nullptr, nullptr, nullptr, nullptr, Rk, nullptr);
    }
    // K3: attention (LDS-resident score rows, ~196KB dynamic LDS per WG)
    {
        size_t smem = (size_t)16 * KLEN * sizeof(float)      // S
                    + (size_t)16 * KLEN * sizeof(bf16_t)     // Braw / P
                    + 64 * sizeof(float);                    // rowinv + pad
        (void)hipFuncSetAttribute((const void*)attn_kernel,
                                  hipFuncAttributeMaxDynamicSharedMemorySize,
                                  (int)smem);
        dim3 grid(QLEN / 16, N_HEAD, BSZ);
        attn_kernel<<<grid, 128, smem, stream>>>(Qw, Qr, Kb, Vt, Rk, attnbf);
    }
    // K4: output projection  [4096,1024] @ [1024,1024]^T -> f32 d_out
    {
        dim3 grid(D_MODEL / 128, (QLEN * BSZ) / 64);
        wmma_gemm_kernel<<<grid, 256, 0, stream>>>(attnbf, owbf,
            QLEN * BSZ, D_MODEL, D_MODEL, EP_F32,
            nullptr, nullptr, nullptr, nullptr, nullptr, nullptr, nullptr, out);
    }
}
